// SimpleRNN_59055800320032
// MI455X (gfx1250) — compile-verified, matrix operand order checked by reading, not run on hardware
//
#include <hip/hip_runtime.h>
#include <hip/hip_bf16.h>
#include <math.h>

// Problem constants (from reference): B=16384, T=28, I=28, H=128, L=2, C=10
#define BATCH 16384
#define TSEQ  28
#define IDIM  28
#define HDIM  128
#define CDIM  10

#define WAVES          4
#define ROWS_PER_WAVE  16
#define ROWS_PER_BLOCK (WAVES * ROWS_PER_WAVE)

typedef __attribute__((ext_vector_type(16))) __bf16 v16bf;
typedef __attribute__((ext_vector_type(8)))  __bf16 bf16x8;
typedef __attribute__((ext_vector_type(8)))  float  v8f;
typedef __attribute__((ext_vector_type(4)))  float  f32x4;

static __device__ __forceinline__ v8f wmma_bf16(v16bf a, v16bf b, v8f c) {
    // D = A(16x32 bf16) * B(32x16 bf16) + C(16x16 f32)
    return __builtin_amdgcn_wmma_f32_16x16x32_bf16(
        /*neg_a=*/false, a, /*neg_b=*/false, b,
        /*c_mod=*/(short)0, c, /*reuse_a=*/false, /*reuse_b=*/false);
}

// Branch-free tanh. gfx1250 has native V_TANH_F32; otherwise use
// tanh(x) = 1 - 2/(exp2(2*log2e*x)+1), which saturates to +/-1 at both
// ends with zero compares (keeps EXEC untouched in the WMMA hot loop).
static __device__ __forceinline__ float fast_tanh(float v) {
#if __has_builtin(__builtin_amdgcn_tanhf)
    return __builtin_amdgcn_tanhf(v);
#elif __has_builtin(__builtin_amdgcn_tanh_f32)
    return __builtin_amdgcn_tanh_f32(v);
#else
    float e = __builtin_amdgcn_exp2f(v * 2.885390081777927f);  // 2/ln(2)
    return 1.0f - 2.0f * __builtin_amdgcn_rcpf(e + 1.0f);
#endif
}

// Load one 16x32 (A) or 32x16-as-rows-of-W (B) bf16 fragment from LDS.
// `rows` points at the first of 16 rows, each of leading dimension `ld` (bf16
// elements, row-major along K). Per the CDNA5 ISA layout, lane L holds two
// contiguous 8-element K-chunks at kb+8*hi and kb+16+8*hi (hi = L>>4).
static __device__ __forceinline__ v16bf lds_frag(const __bf16* rows, int ld,
                                                 int lane, int kb) {
    int r  = lane & 15;
    int hi = lane >> 4;
    const __bf16* p = rows + r * ld + kb + 8 * hi;
    bf16x8 lo = *(const bf16x8*)p;
    bf16x8 hv = *(const bf16x8*)(p + 16);
    return __builtin_shufflevector(lo, hv, 0, 1, 2, 3, 4, 5, 6, 7,
                                   8, 9, 10, 11, 12, 13, 14, 15);
}

__global__ __launch_bounds__(WAVES * 32, 1)
void rnn2_fused_wmma(const float* __restrict__ x,
                     const float* __restrict__ Wih0, const float* __restrict__ Whh0,
                     const float* __restrict__ bih0, const float* __restrict__ bhh0,
                     const float* __restrict__ Wih1, const float* __restrict__ Whh1,
                     const float* __restrict__ bih1, const float* __restrict__ bhh1,
                     const float* __restrict__ Wfc,  const float* __restrict__ bfc,
                     float* __restrict__ out) {
    // ---- LDS: one bf16 copy of all weights per workgroup + per-wave h state
    __shared__ __bf16 sWih0[HDIM * 32];         //  8 KB (K padded 28->32 w/ zeros)
    __shared__ __bf16 sWhh0[HDIM * HDIM];       // 32 KB
    __shared__ __bf16 sWih1[HDIM * HDIM];       // 32 KB
    __shared__ __bf16 sWhh1[HDIM * HDIM];       // 32 KB
    __shared__ __bf16 sWfc[16 * HDIM];          //  4 KB (rows padded 10->16)
    __shared__ float  sB0[HDIM], sB1[HDIM], sBfc[16];
    __shared__ __bf16 sH0[WAVES][ROWS_PER_WAVE * HDIM];  // 16 KB
    __shared__ __bf16 sH1[WAVES][ROWS_PER_WAVE * HDIM];  // 16 KB

    const int tid = threadIdx.x;

    // ---- Stage weights (fp32 global -> bf16 LDS), cooperative across block
    for (int i = tid; i < HDIM * 32; i += WAVES * 32) {
        int n = i >> 5, k = i & 31;
        sWih0[i] = (k < IDIM) ? (__bf16)Wih0[n * IDIM + k] : (__bf16)0.0f;
    }
    for (int i = tid; i < HDIM * HDIM; i += WAVES * 32) {
        sWhh0[i] = (__bf16)Whh0[i];
        sWih1[i] = (__bf16)Wih1[i];
        sWhh1[i] = (__bf16)Whh1[i];
    }
    for (int i = tid; i < 16 * HDIM; i += WAVES * 32) {
        int c = i >> 7, k = i & 127;
        sWfc[i] = (c < CDIM) ? (__bf16)Wfc[c * HDIM + k] : (__bf16)0.0f;
    }
    if (tid < HDIM) {
        sB0[tid] = bih0[tid] + bhh0[tid];
        sB1[tid] = bih1[tid] + bhh1[tid];
    }
    if (tid < 16) sBfc[tid] = (tid < CDIM) ? bfc[tid] : 0.0f;
    for (int i = tid; i < WAVES * ROWS_PER_WAVE * HDIM; i += WAVES * 32) {
        (&sH0[0][0])[i] = (__bf16)0.0f;
        (&sH1[0][0])[i] = (__bf16)0.0f;
    }
    __syncthreads();

    const int wid  = tid >> 5;
    const int lane = tid & 31;
    const int m    = lane & 15;   // row within tile (A) / col (C frag)
    const int hi   = lane >> 4;
    const int b0   = blockIdx.x * ROWS_PER_BLOCK + wid * ROWS_PER_WAVE;

    __bf16* h0 = sH0[wid];
    __bf16* h1 = sH1[wid];

    v8f nh[8];  // per-step new hidden state, 8 N-tiles of v8f (64 VGPRs)

    for (int t = 0; t < TSEQ; ++t) {
        // ---- x A-fragment (16x28 fp32 -> 16x32 bf16, tail zero-padded)
        const float* row = x + ((size_t)(b0 + m) * TSEQ + t) * IDIM;
        if (t + 1 < TSEQ) __builtin_prefetch(row + IDIM, 0, 0);
        f32x4 zero4 = {0.f, 0.f, 0.f, 0.f};
        f32x4 a0 = *(const f32x4*)(row + 8 * hi);            // K = 8hi .. 8hi+3
        f32x4 a1 = *(const f32x4*)(row + 8 * hi + 4);        // K = 8hi+4 .. 8hi+7
        f32x4 a2 = *(const f32x4*)(row + 16 + 8 * hi);       // K = 16+8hi .. +3
        f32x4 a3 = hi ? zero4 : *(const f32x4*)(row + 20);   // K 28..31 -> 0
        v16bf ax;
        ax[0]  = (__bf16)a0[0]; ax[1]  = (__bf16)a0[1];
        ax[2]  = (__bf16)a0[2]; ax[3]  = (__bf16)a0[3];
        ax[4]  = (__bf16)a1[0]; ax[5]  = (__bf16)a1[1];
        ax[6]  = (__bf16)a1[2]; ax[7]  = (__bf16)a1[3];
        ax[8]  = (__bf16)a2[0]; ax[9]  = (__bf16)a2[1];
        ax[10] = (__bf16)a2[2]; ax[11] = (__bf16)a2[3];
        ax[12] = (__bf16)a3[0]; ax[13] = (__bf16)a3[1];
        ax[14] = (__bf16)a3[2]; ax[15] = (__bf16)a3[3];

        // ---- Layer 0: h0 = tanh(x Wih0^T + h0 Whh0^T + b0)
        v16bf hp[4];
#pragma unroll
        for (int ks = 0; ks < 4; ++ks) hp[ks] = lds_frag(h0, HDIM, lane, ks * 32);
#pragma unroll
        for (int nt = 0; nt < 8; ++nt) {
            float bb = sB0[nt * 16 + m];
            v8f acc;  // bias folded into C init (same column across all 8 rows)
#pragma unroll
            for (int r = 0; r < 8; ++r) acc[r] = bb;
            acc = wmma_bf16(ax, lds_frag(sWih0 + nt * 16 * 32, 32, lane, 0), acc);
#pragma unroll
            for (int ks = 0; ks < 4; ++ks)
                acc = wmma_bf16(hp[ks],
                                lds_frag(sWhh0 + nt * 16 * HDIM, HDIM, lane, ks * 32),
                                acc);
#pragma unroll
            for (int r = 0; r < 8; ++r) acc[r] = fast_tanh(acc[r]);
            nh[nt] = acc;
        }
        // all reads of old h0 done above; commit new h0 (bf16, row-major [16][128])
#pragma unroll
        for (int nt = 0; nt < 8; ++nt)
#pragma unroll
            for (int r = 0; r < 8; ++r)
                h0[(r + 8 * hi) * HDIM + nt * 16 + m] = (__bf16)nh[nt][r];

        // ---- Layer 1: h1 = tanh(h0 Wih1^T + h1 Whh1^T + b1)
        v16bf h0n[4], h1p[4];
#pragma unroll
        for (int ks = 0; ks < 4; ++ks) {
            h0n[ks] = lds_frag(h0, HDIM, lane, ks * 32);
            h1p[ks] = lds_frag(h1, HDIM, lane, ks * 32);
        }
#pragma unroll
        for (int nt = 0; nt < 8; ++nt) {
            float bb = sB1[nt * 16 + m];
            v8f acc;
#pragma unroll
            for (int r = 0; r < 8; ++r) acc[r] = bb;
#pragma unroll
            for (int ks = 0; ks < 4; ++ks) {
                acc = wmma_bf16(h0n[ks],
                                lds_frag(sWih1 + nt * 16 * HDIM, HDIM, lane, ks * 32),
                                acc);
                acc = wmma_bf16(h1p[ks],
                                lds_frag(sWhh1 + nt * 16 * HDIM, HDIM, lane, ks * 32),
                                acc);
            }
#pragma unroll
            for (int r = 0; r < 8; ++r) acc[r] = fast_tanh(acc[r]);
            nh[nt] = acc;
        }
#pragma unroll
        for (int nt = 0; nt < 8; ++nt)
#pragma unroll
            for (int r = 0; r < 8; ++r)
                h1[(r + 8 * hi) * HDIM + nt * 16 + m] = (__bf16)nh[nt][r];
    }

    // ---- FC head on last h1: out = h1 Wfc^T + bfc (N padded 10->16)
    v8f acc = {};
#pragma unroll
    for (int ks = 0; ks < 4; ++ks)
        acc = wmma_bf16(lds_frag(h1, HDIM, lane, ks * 32),
                        lds_frag(sWfc, HDIM, lane, ks * 32), acc);
    if (m < CDIM) {
        float bb = sBfc[m];
#pragma unroll
        for (int r = 0; r < 8; ++r)
            out[(size_t)(b0 + r + 8 * hi) * CDIM + m] = acc[r] + bb;
    }
}

extern "C" void kernel_launch(void* const* d_in, const int* in_sizes, int n_in,
                              void* d_out, int out_size, void* d_ws, size_t ws_size,
                              hipStream_t stream) {
    const float* x    = (const float*)d_in[0];
    const float* Wih0 = (const float*)d_in[1];
    const float* Whh0 = (const float*)d_in[2];
    const float* bih0 = (const float*)d_in[3];
    const float* bhh0 = (const float*)d_in[4];
    const float* Wih1 = (const float*)d_in[5];
    const float* Whh1 = (const float*)d_in[6];
    const float* bih1 = (const float*)d_in[7];
    const float* bhh1 = (const float*)d_in[8];
    const float* Wfc  = (const float*)d_in[9];
    const float* bfc  = (const float*)d_in[10];
    float* out = (float*)d_out;

    dim3 grid(BATCH / ROWS_PER_BLOCK);  // 16384/64 = 256 workgroups
    dim3 block(WAVES * 32);             // 4 wave32 per workgroup
    rnn2_fused_wmma<<<grid, block, 0, stream>>>(
        x, Wih0, Whh0, bih0, bhh0, Wih1, Whh1, bih1, bhh1, Wfc, bfc, out);
}
